// ScaledDotProductAttention_62448824484005
// MI455X (gfx1250) — matrix operand order repaired and checked by reading, not compiled, not run on hardware
//
#include <hip/hip_runtime.h>

typedef __attribute__((ext_vector_type(16))) _Float16 v16h;
typedef __attribute__((ext_vector_type(8)))  _Float16 v8h;
typedef __attribute__((ext_vector_type(8)))  float    v8f;
typedef int v4i __attribute__((vector_size(16)));   // matches builtin param type

// Problem sizes (from reference setup_inputs)
#define B_  16
#define S_  2048
#define D_  128

constexpr int QTILE  = 64;    // query rows per workgroup (16 per wave)
constexpr int KTILE  = 64;    // key rows per iteration
constexpr int NWAVES = 4;     // 128 threads, wave32
constexpr int KP = D_ + 8;    // Klds row stride (f16): 272B rows, 16B aligned
constexpr int VP = KTILE + 8; // Vtlds row stride: 144B rows, 16B aligned
constexpr int PP = KTILE + 8; // Plds row stride

// ---- Async global->LDS (CDNA5, ASYNCcnt-tracked), with safe fallback. ----
#if __has_builtin(__builtin_amdgcn_global_load_async_to_lds_b128)
#define ASYNC_G2L 1
typedef __attribute__((address_space(1))) v4i* gv4i_p;  // global int4*
typedef __attribute__((address_space(3))) v4i* lv4i_p;  // LDS int4*
#else
#define ASYNC_G2L 0
#endif

__device__ __forceinline__ void g2l_b128(const float* g, float* l) {
#if ASYNC_G2L
    __builtin_amdgcn_global_load_async_to_lds_b128((gv4i_p)(g), (lv4i_p)(l), 0, 0);
#else
    *(float4*)l = *(const float4*)g;   // synchronous fallback
#endif
}

__device__ __forceinline__ void wait_async0() {
#if ASYNC_G2L
#if __has_builtin(__builtin_amdgcn_s_wait_asynccnt)
    __builtin_amdgcn_s_wait_asynccnt(0);
#else
    asm volatile("s_wait_asynccnt 0x0" ::: "memory");
#endif
#endif
}

// Load a 16-wide f16 B-fragment (32x16 WMMA B layout) from contiguous LDS.
__device__ __forceinline__ v16h lds_bfrag(const _Float16* p) {
    const v8h* pv = (const v8h*)p;
    v8h lo = pv[0];
    v8h hi = pv[1];
    v16h r;
#pragma unroll
    for (int i = 0; i < 8; ++i) { r[i] = lo[i]; r[8 + i] = hi[i]; }
    return r;
}

__global__ __launch_bounds__(NWAVES * 32)
void fa_fwd_kernel(const float* __restrict__ Q, const float* __restrict__ K,
                   const float* __restrict__ V, float* __restrict__ O)
{
    // Raw f32 staging tiles (async DMA destination) + converted f16 tiles.
    __shared__ __align__(16) float    rawK[KTILE * D_];       // 32 KB
    __shared__ __align__(16) float    rawV[KTILE * D_];       // 32 KB
    __shared__ __align__(16) _Float16 Klds[KTILE][KP];        // [key][d]
    __shared__ __align__(16) _Float16 Vtlds[D_][VP];          // [d][key]
    __shared__ __align__(16) _Float16 Plds[NWAVES][16][PP];   // per-wave P

    const int tid  = threadIdx.x;
    const int wave = tid >> 5;
    const int lane = tid & 31;
    const int half = lane >> 4;   // wave32: lanes 0-15 vs 16-31
    const int lq   = lane & 15;

    const int b  = blockIdx.x / (S_ / QTILE);
    const int qt = blockIdx.x % (S_ / QTILE);
    const int q0 = qt * QTILE;

    const float scale = 0.08838834764831845f; // 1/sqrt(128)

    const float* Kbase = K + (size_t)b * S_ * D_;
    const float* Vbase = V + (size_t)b * S_ * D_;

    // ---- Gather Q A-fragments (16x32 f16 WMMA A layout), rows q0+wave*16+m.
    v16h qa[4];
    {
        const float* qrow = Q + ((size_t)b * S_ + (size_t)(q0 + wave * 16 + lq)) * D_;
#pragma unroll
        for (int kc = 0; kc < 4; ++kc) {
#pragma unroll
            for (int vg = 0; vg < 8; ++vg) {
                const int k0 = half * 16 + 2 * vg;
                const float2 f = *(const float2*)(qrow + kc * 32 + k0);
                qa[kc][2 * vg]     = (_Float16)f.x;
                qa[kc][2 * vg + 1] = (_Float16)f.y;
            }
        }
    }

    // ---- Online-softmax state (rows m = r + 8*half) and O accumulators.
    float rmax[8], lsum[8];
    v8f   oacc[8];
    const v8f vzero = {};
#pragma unroll
    for (int r = 0; r < 8; ++r) { rmax[r] = -__builtin_inff(); lsum[r] = 0.0f; }
#pragma unroll
    for (int dt = 0; dt < 8; ++dt) oacc[dt] = vzero;

    // ---- Prologue: async-stage raw f32 tile 0.
#pragma unroll
    for (int i = 0; i < 16; ++i) {
        const int lin = (tid + i * (NWAVES * 32)) * 4;   // float index, 16B chunks
        g2l_b128(Kbase + lin, &rawK[lin]);
        g2l_b128(Vbase + lin, &rawV[lin]);
    }
    wait_async0();
    __syncthreads();

    // ---- Key-tile loop (raw tile for kt0 is resident at loop top).
    for (int kt0 = 0; kt0 < S_; kt0 += KTILE) {
        // Convert raw f32 -> f16 K-tile (row major) and V-tile (transposed).
#pragma unroll
        for (int i = 0; i < 16; ++i) {
            const int lin = tid + i * (NWAVES * 32);
            const int row = lin >> 5;      // 32 float4 per 128-elem row
            const int c4  = lin & 31;
            const float4 fk = *(const float4*)&rawK[lin * 4];
            _Float16* dk = &Klds[row][c4 * 4];
            dk[0] = (_Float16)fk.x; dk[1] = (_Float16)fk.y;
            dk[2] = (_Float16)fk.z; dk[3] = (_Float16)fk.w;
            const float4 fv = *(const float4*)&rawV[lin * 4];
            Vtlds[c4 * 4 + 0][row] = (_Float16)fv.x;
            Vtlds[c4 * 4 + 1][row] = (_Float16)fv.y;
            Vtlds[c4 * 4 + 2][row] = (_Float16)fv.z;
            Vtlds[c4 * 4 + 3][row] = (_Float16)fv.w;
        }
        __syncthreads();   // f16 tiles ready; raw buffers free for next DMA

        // Kick off async DMA of the NEXT raw tile; overlaps the math below.
        if (kt0 + KTILE < S_) {
            const float* Kg = Kbase + (size_t)(kt0 + KTILE) * D_;
            const float* Vg = Vbase + (size_t)(kt0 + KTILE) * D_;
#pragma unroll
            for (int i = 0; i < 16; ++i) {
                const int lin = (tid + i * (NWAVES * 32)) * 4;
                g2l_b128(Kg + lin, &rawK[lin]);
                g2l_b128(Vg + lin, &rawV[lin]);
            }
        }

        // ---- S = Q @ K^T : 4 N-tiles of 16 keys, K-dim 128 in 4 chunks of 32.
        v8f accs[4];
#pragma unroll
        for (int nt = 0; nt < 4; ++nt) {
            v8f c = {};
#pragma unroll
            for (int kc = 0; kc < 4; ++kc) {
                const v16h bf = lds_bfrag(&Klds[nt * 16 + lq][kc * 32 + half * 16]);
                c = __builtin_amdgcn_wmma_f32_16x16x32_f16(
                        false, qa[kc], false, bf, (short)0, c, false, false);
            }
            accs[nt] = c;
        }

        // ---- Online softmax over this 16x64 score block.
#pragma unroll
        for (int nt = 0; nt < 4; ++nt)
#pragma unroll
            for (int r = 0; r < 8; ++r) accs[nt][r] *= scale;

#pragma unroll
        for (int r = 0; r < 8; ++r) {
            float tm = fmaxf(fmaxf(accs[0][r], accs[1][r]),
                             fmaxf(accs[2][r], accs[3][r]));
#pragma unroll
            for (int off = 8; off >= 1; off >>= 1)
                tm = fmaxf(tm, __shfl_xor(tm, off, 32)); // within 16-lane half
            const float nm    = fmaxf(rmax[r], tm);
            const float alpha = __expf(rmax[r] - nm);
            rmax[r] = nm;
            float rs = 0.0f;
#pragma unroll
            for (int nt = 0; nt < 4; ++nt) {
                const float e = __expf(accs[nt][r] - nm);
                accs[nt][r] = e;
                rs += e;
            }
#pragma unroll
            for (int off = 8; off >= 1; off >>= 1)
                rs += __shfl_xor(rs, off, 32);
            lsum[r] = lsum[r] * alpha + rs;
#pragma unroll
            for (int dt = 0; dt < 8; ++dt) oacc[dt][r] *= alpha;
        }

        // ---- Spill P (f16) to per-wave LDS region (C layout -> row major).
#pragma unroll
        for (int nt = 0; nt < 4; ++nt)
#pragma unroll
            for (int r = 0; r < 8; ++r)
                Plds[wave][r + half * 8][nt * 16 + lq] = (_Float16)accs[nt][r];
        __syncthreads();

        // ---- Re-gather P as A-fragments (16x64 -> two 16x32 K-steps).
        v16h pa[2];
#pragma unroll
        for (int kti = 0; kti < 2; ++kti) {
            const _Float16* prow = &Plds[wave][lq][kti * 32];
#pragma unroll
            for (int vg = 0; vg < 8; ++vg) {
                const int k0 = half * 16 + 2 * vg;
                pa[kti][2 * vg]     = prow[k0];
                pa[kti][2 * vg + 1] = prow[k0 + 1];
            }
        }

        // ---- O += P @ V : 8 D-tiles x 2 K-steps of 32 keys.
#pragma unroll
        for (int dt = 0; dt < 8; ++dt) {
            v8f c = oacc[dt];
#pragma unroll
            for (int kti = 0; kti < 2; ++kti) {
                const v16h bf = lds_bfrag(&Vtlds[dt * 16 + lq][kti * 32 + half * 16]);
                c = __builtin_amdgcn_wmma_f32_16x16x32_f16(
                        false, pa[kti], false, bf, (short)0, c, false, false);
            }
            oacc[dt] = c;
        }

        // Next raw tile must be fully landed before the convert pass reads it.
        wait_async0();
        __syncthreads();
    }

    // ---- Epilogue: normalize by running row sum, store f32 output.
#pragma unroll
    for (int r = 0; r < 8; ++r) {
        const float inv = 1.0f / lsum[r];
        const int   m   = r + half * 8;
        float* orow = O + ((size_t)b * S_ + (size_t)(q0 + wave * 16 + m)) * D_;
#pragma unroll
        for (int dt = 0; dt < 8; ++dt)
            orow[dt * 16 + lq] = oacc[dt][r] * inv;
    }
}

extern "C" void kernel_launch(void* const* d_in, const int* in_sizes, int n_in,
                              void* d_out, int out_size, void* d_ws, size_t ws_size,
                              hipStream_t stream) {
    const float* q = (const float*)d_in[0];
    const float* k = (const float*)d_in[1];
    const float* v = (const float*)d_in[2];
    float* o = (float*)d_out;
    (void)in_sizes; (void)n_in; (void)out_size; (void)d_ws; (void)ws_size;

    dim3 grid(B_ * (S_ / QTILE));   // 16 batches x 32 query tiles = 512 WGs
    dim3 block(NWAVES * 32);        // 4 waves, wave32
    hipLaunchKernelGGL(fa_fwd_kernel, grid, block, 0, stream, q, k, v, o);
}
